// MaskedCrossAttention_5669356835458
// MI455X (gfx1250) — matrix operand order repaired and checked by reading, not compiled, not run on hardware
//
#include <hip/hip_runtime.h>
#include <float.h>

// ---------------------------------------------------------------------------
// Masked cross attention (Flamingo-style) for MI455X / gfx1250, wave32 WMMA.
// All GEMMs + attention use v_wmma_f32_16x16x32_f16 (f32 accumulate).
// GEMMs are register-blocked 16(M) x 64(N) per wave: one A fragment feeds
// four B fragments / four accumulators per 32-deep K step.
// ---------------------------------------------------------------------------

typedef _Float16 half_t;
typedef __attribute__((ext_vector_type(8)))  half_t v8h;
typedef __attribute__((ext_vector_type(16))) half_t v16h;
typedef __attribute__((ext_vector_type(8)))  float  v8f;

#define NB_B     2
#define NB_T     2048      // text tokens
#define NB_DIM   1024
#define NB_MTOK  256       // tokens per media block
#define NB_HEADS 16
#define NB_DHEAD 64
#define NB_INNER 1024
#define NB_TM    2048      // total media tokens (8 * 256)
#define NB_SCALE 0.125f    // DIM_HEAD ** -0.5

__device__ __forceinline__ v16h cat16(v8h lo, v8h hi) {
  return __builtin_shufflevector(lo, hi, 0,1,2,3,4,5,6,7,8,9,10,11,12,13,14,15);
}
__device__ __forceinline__ v8f wmma16(v16h a, v16h b, v8f c) {
  return __builtin_amdgcn_wmma_f32_16x16x32_f16(false, a, false, b, (short)0, c,
                                                false, false);
}

// ---------------------------------------------------------------------------
// Convert fp32 weight [K x N] to f16 transposed [N x K] (contiguous-K rows so
// WMMA B-fragments are contiguous per lane).
// ---------------------------------------------------------------------------
__global__ void cvt_w_kernel(const float* __restrict__ W, half_t* __restrict__ Wt,
                             int K, int N) {
  int total = K * N;
  for (int i = blockIdx.x * blockDim.x + threadIdx.x; i < total;
       i += gridDim.x * blockDim.x) {
    int k = i / N, n = i - k * N;
    Wt[(size_t)n * K + k] = (half_t)W[i];
  }
}

__global__ void cvt_f16_kernel(const float* __restrict__ X, half_t* __restrict__ Y,
                               int total) {
  for (int i = blockIdx.x * blockDim.x + threadIdx.x; i < total;
       i += gridDim.x * blockDim.x)
    Y[i] = (half_t)X[i];
}

// ---------------------------------------------------------------------------
// LayerNorm(1024) -> f16, one block per row.
// ---------------------------------------------------------------------------
__global__ __launch_bounds__(256)
void ln_kernel(const float* __restrict__ x, const float* __restrict__ g,
               const float* __restrict__ be, half_t* __restrict__ y) {
  __shared__ float red[256];
  const int row = blockIdx.x;
  const float* xr = x + (size_t)row * NB_DIM;
  float s = 0.f, s2 = 0.f;
  for (int i = threadIdx.x; i < NB_DIM; i += 256) {
    float v = xr[i]; s += v; s2 += v * v;
  }
  red[threadIdx.x] = s; __syncthreads();
  for (int w = 128; w > 0; w >>= 1) {
    if ((int)threadIdx.x < w) red[threadIdx.x] += red[threadIdx.x + w];
    __syncthreads();
  }
  float mean = red[0] * (1.f / NB_DIM);
  __syncthreads();
  red[threadIdx.x] = s2; __syncthreads();
  for (int w = 128; w > 0; w >>= 1) {
    if ((int)threadIdx.x < w) red[threadIdx.x] += red[threadIdx.x + w];
    __syncthreads();
  }
  float var = red[0] * (1.f / NB_DIM) - mean * mean;
  float rs = rsqrtf(var + 1e-5f);
  half_t* yr = y + (size_t)row * NB_DIM;
  for (int i = threadIdx.x; i < NB_DIM; i += 256)
    yr[i] = (half_t)(((xr[i] - mean) * rs) * g[i] + be[i]);
}

// ---------------------------------------------------------------------------
// text_time = cumsum(media_locations) per batch row (tiny, serial).
// ---------------------------------------------------------------------------
__global__ void tt_kernel(const unsigned char* __restrict__ locs,
                          int* __restrict__ tt, int T) {
  if (threadIdx.x == 0) {
    int b = blockIdx.x, c = 0;
    for (int i = 0; i < T; ++i) {
      c += locs[(size_t)b * T + i] ? 1 : 0;
      tt[(size_t)b * T + i] = c;
    }
  }
}

// ---------------------------------------------------------------------------
// 16(M) x 64(N) register-blocked tile per wave.  A: [M x K] row-major f16.
// Bt: [N x K] row-major f16 (original B transposed).  Fragment loads follow
// one contiguous K order for A (lane m=L&15, K 16*(L>>4)+{0..15})
// and B (lane n=L&15, K halves 16*(L>>4)+{0..15}); both contiguous.
// One A fragment is reused across four B fragments (4 wmma per K-step).
// ---------------------------------------------------------------------------
__device__ __forceinline__ void wmma_tile4(const half_t* __restrict__ A,
                                           const half_t* __restrict__ Bt,
                                           int tm, int tn, int K, int ln, int hi,
                                           v8f c[4]) {
  const half_t* ar = A  + (size_t)(tm + ln) * K;
  const half_t* br = Bt + (size_t)(tn + ln) * K;
  const size_t bstride = (size_t)16 * K;          // next 16-col B group
  for (int kc = 0; kc < K; kc += 32) {
    __builtin_prefetch(ar + kc + 256, 0, 1);
    __builtin_prefetch(br + kc + 256, 0, 1);
    v16h a = cat16(*(const v8h*)(ar + kc + 16 * hi),
                   *(const v8h*)(ar + kc + 16 * hi + 8));
#pragma unroll
    for (int t = 0; t < 4; ++t) {
      const half_t* bp = br + (size_t)t * bstride + kc + 16 * hi;
      c[t] = wmma16(a, cat16(*(const v8h*)bp, *(const v8h*)(bp + 8)), c[t]);
    }
  }
}

// Q = LN(x) @ Wq, pre-scaled by DIM_HEAD^-0.5, stored f16 [B*T x INNER].
__global__ __launch_bounds__(256)
void gemm_q_kernel(const half_t* __restrict__ A, const half_t* __restrict__ Bt,
                   half_t* __restrict__ Q, int M, int N, int K, float scale) {
  int tile = blockIdx.x * (blockDim.x >> 5) + ((int)threadIdx.x >> 5);
  int tilesN = N >> 6;                             // 64-wide N tiles
  if (tile >= (M >> 4) * tilesN) return;
  int tm = (tile / tilesN) << 4, tn = (tile % tilesN) << 6;
  int lane = threadIdx.x & 31, ln = lane & 15, hi = lane >> 4;
  v8f c[4] = {{}, {}, {}, {}};
  wmma_tile4(A, Bt, tm, tn, K, ln, hi, c);
#pragma unroll
  for (int t = 0; t < 4; ++t)
#pragma unroll
    for (int r = 0; r < 8; ++r)
      Q[(size_t)(tm + 8 * hi + r) * N + tn + t * 16 + ln] = (half_t)(c[t][r] * scale);
}

// KV = media @ Wkv.  Cols [0,1024) -> K  (natural [B*TM x INNER] f16);
// cols [1024,2048) -> V stored TRANSPOSED per head: Vt[b][h][d][tm] so the
// P·V WMMA B-fragments are contiguous along the key dimension.
__global__ __launch_bounds__(256)
void gemm_kv_kernel(const half_t* __restrict__ A, const half_t* __restrict__ Bt,
                    half_t* __restrict__ Kout, half_t* __restrict__ Vt,
                    int M, int N, int K) {
  int tile = blockIdx.x * (blockDim.x >> 5) + ((int)threadIdx.x >> 5);
  int tilesN = N >> 6;
  if (tile >= (M >> 4) * tilesN) return;
  int tm = (tile / tilesN) << 4, tn = (tile % tilesN) << 6;
  int lane = threadIdx.x & 31, ln = lane & 15, hi = lane >> 4;
  v8f c[4] = {{}, {}, {}, {}};
  wmma_tile4(A, Bt, tm, tn, K, ln, hi, c);
#pragma unroll
  for (int t = 0; t < 4; ++t) {
    int col = tn + t * 16 + ln;
#pragma unroll
    for (int r = 0; r < 8; ++r) {
      int row = tm + 8 * hi + r;
      float v = c[t][r];
      if (col < NB_INNER) {
        Kout[(size_t)row * NB_INNER + col] = (half_t)v;
      } else {
        int nv = col - NB_INNER, hh = nv >> 6, dd = nv & 63;
        int bb = row >> 11, tk = row & (NB_TM - 1);
        Vt[(((size_t)bb * NB_HEADS + hh) * NB_DHEAD + dd) * NB_TM + tk] = (half_t)v;
      }
    }
  }
}

// out = attn_out @ Wout -> fp32 result.
__global__ __launch_bounds__(256)
void gemm_out_kernel(const half_t* __restrict__ A, const half_t* __restrict__ Bt,
                     float* __restrict__ O, int M, int N, int K) {
  int tile = blockIdx.x * (blockDim.x >> 5) + ((int)threadIdx.x >> 5);
  int tilesN = N >> 6;
  if (tile >= (M >> 4) * tilesN) return;
  int tm = (tile / tilesN) << 4, tn = (tile % tilesN) << 6;
  int lane = threadIdx.x & 31, ln = lane & 15, hi = lane >> 4;
  v8f c[4] = {{}, {}, {}, {}};
  wmma_tile4(A, Bt, tm, tn, K, ln, hi, c);
#pragma unroll
  for (int t = 0; t < 4; ++t)
#pragma unroll
    for (int r = 0; r < 8; ++r)
      O[(size_t)(tm + 8 * hi + r) * N + tn + t * 16 + ln] = c[t][r];
}

// ---------------------------------------------------------------------------
// Masked flash attention.  One wave per (b, h, 16-query tile).  text_time is
// monotone so the tile needs only media blocks j in [max(1,tt_min), tt_max]
// (almost always exactly one 256-key block).  Online softmax with width-16
// shuffle reductions (D-layout rows live in a fixed 16-lane half); P goes
// D-layout -> LDS -> A-layout for the P·V WMMA.
// ---------------------------------------------------------------------------
__global__ __launch_bounds__(32)
void attn_kernel(const half_t* __restrict__ Qf, const half_t* __restrict__ Kf,
                 const half_t* __restrict__ Vt, const int* __restrict__ ttime,
                 half_t* __restrict__ O) {
  __shared__ __align__(32) half_t pLds[16 * 32];
  const int QT = NB_T / 16;
  const int blk = blockIdx.x;
  const int qt = blk % QT;
  const int h  = (blk / QT) % NB_HEADS;
  const int b  = blk / (QT * NB_HEADS);
  const int lane = (int)threadIdx.x & 31, ln = lane & 15, hi = lane >> 4;
  const int i0 = qt * 16;

  // Pack the tile's 16 text_time values (<=8, fits 4 bits) in a scalar.
  unsigned long long ttp = 0ull;
  int ttmin = 1 << 30, ttmax = 0;
#pragma unroll
  for (int m = 0; m < 16; ++m) {
    int t = ttime[b * NB_T + i0 + m];
    ttp |= (unsigned long long)(t & 15) << (4 * m);
    ttmin = t < ttmin ? t : ttmin;
    ttmax = t > ttmax ? t : ttmax;
  }

  // Q tile as two A-fragments (d 0..31, d 32..63); Q is pre-scaled.
  const half_t* qrow = Qf + ((size_t)(b * NB_T + i0 + ln)) * NB_INNER + h * NB_DHEAD;
  v16h A0 = cat16(*(const v8h*)(qrow + 16 * hi),      *(const v8h*)(qrow + 8 + 16 * hi));
  v16h A1 = cat16(*(const v8h*)(qrow + 32 + 16 * hi), *(const v8h*)(qrow + 40 + 16 * hi));

  v8f acc0 = {}, acc1 = {}, acc2 = {}, acc3 = {};
  float mrow[8], lrow[8];
#pragma unroll
  for (int r = 0; r < 8; ++r) { mrow[r] = -3.0e38f; lrow[r] = 0.f; }

  const int js = ttmin < 1 ? 1 : ttmin;
  for (int j = js; j <= ttmax; ++j) {
    unsigned mb = 0u;
#pragma unroll
    for (int m = 0; m < 16; ++m)
      mb |= (((ttp >> (4 * m)) & 15ull) == (unsigned long long)j) ? (1u << m) : 0u;
    const int kb = (j - 1) * NB_MTOK;

    for (int kt = 0; kt < 16; kt += 2) {          // 2 key-tiles = one 32-key chunk
      const half_t* kr0 = Kf + ((size_t)(b * NB_TM + kb + kt * 16 + ln)) * NB_INNER
                             + h * NB_DHEAD + 16 * hi;
      const half_t* kr1 = kr0 + (size_t)16 * NB_INNER;
      v8f z0 = {}, z1 = {};
      v8f s0v = wmma16(A0, cat16(*(const v8h*)kr0,        *(const v8h*)(kr0 + 8)),  z0);
      s0v     = wmma16(A1, cat16(*(const v8h*)(kr0 + 32), *(const v8h*)(kr0 + 40)), s0v);
      v8f s1v = wmma16(A0, cat16(*(const v8h*)kr1,        *(const v8h*)(kr1 + 8)),  z1);
      s1v     = wmma16(A1, cat16(*(const v8h*)(kr1 + 32), *(const v8h*)(kr1 + 40)), s1v);

#pragma unroll
      for (int r = 0; r < 8; ++r) {
        const bool ok = (mb >> (r + 8 * hi)) & 1u;
        float s0 = ok ? s0v[r] : -3.0e38f;
        float s1 = ok ? s1v[r] : -3.0e38f;
        float mx = fmaxf(s0, s1);
        mx = fmaxf(mx, __shfl_xor(mx, 1, 16));
        mx = fmaxf(mx, __shfl_xor(mx, 2, 16));
        mx = fmaxf(mx, __shfl_xor(mx, 4, 16));
        mx = fmaxf(mx, __shfl_xor(mx, 8, 16));
        const float newm = fmaxf(mrow[r], mx);
        float scale, p0, p1;
        if (newm <= -1.0e38f) { scale = 1.f; p0 = 0.f; p1 = 0.f; }
        else {
          scale = __expf(mrow[r] - newm);
          p0 = ok ? __expf(s0 - newm) : 0.f;
          p1 = ok ? __expf(s1 - newm) : 0.f;
        }
        float ps = p0 + p1;
        ps += __shfl_xor(ps, 1, 16);
        ps += __shfl_xor(ps, 2, 16);
        ps += __shfl_xor(ps, 4, 16);
        ps += __shfl_xor(ps, 8, 16);
        lrow[r] = lrow[r] * scale + ps;
        mrow[r] = newm;
        acc0[r] *= scale; acc1[r] *= scale; acc2[r] *= scale; acc3[r] *= scale;
        const int m = r + 8 * hi;
        pLds[m * 32 + ln]      = (half_t)p0;
        pLds[m * 32 + 16 + ln] = (half_t)p1;
      }
      __syncthreads();   // single-wave block: orders LDS stores before loads

      const half_t* pr = &pLds[ln * 32 + 16 * hi];
      v16h Ap = cat16(*(const v8h*)pr, *(const v8h*)(pr + 8));
      const half_t* vb = Vt + (((size_t)(b * NB_HEADS + h)) * NB_DHEAD + ln) * NB_TM
                            + kb + kt * 16 + 16 * hi;
      acc0 = wmma16(Ap, cat16(*(const v8h*)vb,                *(const v8h*)(vb + 8)),               acc0);
      acc1 = wmma16(Ap, cat16(*(const v8h*)(vb + 16 * NB_TM), *(const v8h*)(vb + 16 * NB_TM + 8)), acc1);
      acc2 = wmma16(Ap, cat16(*(const v8h*)(vb + 32 * NB_TM), *(const v8h*)(vb + 32 * NB_TM + 8)), acc2);
      acc3 = wmma16(Ap, cat16(*(const v8h*)(vb + 48 * NB_TM), *(const v8h*)(vb + 48 * NB_TM + 8)), acc3);
      __syncthreads();
    }
  }

#pragma unroll
  for (int r = 0; r < 8; ++r) {
    const float inv = lrow[r] > 0.f ? 1.f / lrow[r] : 0.f;  // tt==0 rows -> 0
    const int row = i0 + r + 8 * hi;
    half_t* op = O + ((size_t)(b * NB_T + row)) * NB_INNER + h * NB_DHEAD;
    op[ln]      = (half_t)(acc0[r] * inv);
    op[16 + ln] = (half_t)(acc1[r] * inv);
    op[32 + ln] = (half_t)(acc2[r] * inv);
    op[48 + ln] = (half_t)(acc3[r] * inv);
  }
}

// ---------------------------------------------------------------------------
extern "C" void kernel_launch(void* const* d_in, const int* in_sizes, int n_in,
                              void* d_out, int out_size, void* d_ws, size_t ws_size,
                              hipStream_t stream) {
  (void)in_sizes; (void)n_in; (void)out_size; (void)ws_size;
  const float* x       = (const float*)d_in[0];
  const float* media   = (const float*)d_in[1];
  const unsigned char* locs = (const unsigned char*)d_in[2];   // bool
  const float* gamma   = (const float*)d_in[3];
  const float* beta    = (const float*)d_in[4];
  const float* Wq      = (const float*)d_in[5];
  const float* Wkv     = (const float*)d_in[6];
  const float* Wout    = (const float*)d_in[7];
  float* out = (float*)d_out;

  char* ws = (char*)d_ws;
  size_t off = 0;
  auto take = [&](size_t bytes) {
    char* p = ws + off;
    off = (off + bytes + 255) & ~(size_t)255;
    return p;
  };
  const size_t h2 = sizeof(half_t);
  half_t* xn    = (half_t*)take((size_t)NB_B * NB_T  * NB_DIM * h2);   // 8 MB
  half_t* medh  = (half_t*)take((size_t)NB_B * NB_TM * NB_DIM * h2);   // 8 MB
  half_t* WqT   = (half_t*)take((size_t)NB_DIM * NB_INNER * h2);       // 2 MB
  half_t* WkvT  = (half_t*)take((size_t)NB_DIM * 2 * NB_INNER * h2);   // 4 MB
  half_t* WoutT = (half_t*)take((size_t)NB_INNER * NB_DIM * h2);       // 2 MB
  half_t* Qh    = (half_t*)take((size_t)NB_B * NB_T  * NB_INNER * h2); // 8 MB
  half_t* Kh    = (half_t*)take((size_t)NB_B * NB_TM * NB_INNER * h2); // 8 MB
  half_t* Vth   = (half_t*)take((size_t)NB_B * NB_HEADS * NB_DHEAD * NB_TM * h2); // 8 MB
  half_t* AOh   = (half_t*)take((size_t)NB_B * NB_T  * NB_INNER * h2); // 8 MB
  int*    tt    = (int*)   take((size_t)NB_B * NB_T * sizeof(int));

  cvt_w_kernel<<<1024, 256, 0, stream>>>(Wq,   WqT,   NB_DIM,   NB_INNER);
  cvt_w_kernel<<<2048, 256, 0, stream>>>(Wkv,  WkvT,  NB_DIM,   2 * NB_INNER);
  cvt_w_kernel<<<1024, 256, 0, stream>>>(Wout, WoutT, NB_INNER, NB_DIM);
  cvt_f16_kernel<<<4096, 256, 0, stream>>>(media, medh, NB_B * NB_TM * NB_DIM);
  ln_kernel<<<NB_B * NB_T, 256, 0, stream>>>(x, gamma, beta, xn);
  tt_kernel<<<NB_B, 32, 0, stream>>>(locs, tt, NB_T);

  // 16x64 tiles per wave, 8 waves per 256-thread block.
  const int tilesQ  = (NB_B * NB_T  / 16) * (NB_INNER / 64);      // 4096
  const int tilesKV = (NB_B * NB_TM / 16) * (2 * NB_INNER / 64);  // 8192
  gemm_q_kernel<<<tilesQ / 8, 256, 0, stream>>>(xn, WqT, Qh,
      NB_B * NB_T, NB_INNER, NB_DIM, NB_SCALE);
  gemm_kv_kernel<<<tilesKV / 8, 256, 0, stream>>>(medh, WkvT, Kh, Vth,
      NB_B * NB_TM, 2 * NB_INNER, NB_DIM);
  attn_kernel<<<NB_B * NB_HEADS * (NB_T / 16), 32, 0, stream>>>(Qh, Kh, Vth, tt, AOh);
  gemm_out_kernel<<<tilesQ / 8, 256, 0, stream>>>(AOh, WoutT, out,
      NB_B * NB_T, NB_DIM, NB_INNER);
}
